// I_S_Block_63204738728360
// MI455X (gfx1250) — matrix operand order repaired and checked by reading, not compiled, not run on hardware
//
#include <hip/hip_runtime.h>
#include <hip/hip_bf16.h>

typedef __attribute__((ext_vector_type(16))) _Float16 v16h;
typedef __attribute__((ext_vector_type(8)))  _Float16 v8h;
typedef __attribute__((ext_vector_type(8)))  float    v8f;
typedef __attribute__((ext_vector_type(4)))  float    v4f;

#define Bc  2
#define Sc  2048
#define Hc  768
#define NHc 12
#define HDc 128
#define VDc 64
#define Mc  (Bc*Sc)

#define GEMM_F16OUT 1
#define GEMM_RELU   2

__device__ __forceinline__ v16h cat8(v8h a, v8h b) {
  return __builtin_shufflevector(a, b, 0,1,2,3,4,5,6,7,8,9,10,11,12,13,14,15);
}

// ---------------------------------------------------------------------------
// Generic WMMA GEMM: out = epi(A_f32[M,K] @ W_f32[K,N] + bias (+res) (relu))
// Block tile 128x128, 8 waves (4 x 2), K-step 32, f16 compute / f32 accum.
// Global->LDS staging software-pipelined through registers.
// ---------------------------------------------------------------------------
__global__ __launch_bounds__(256) void gemm_wmma(
    const float* __restrict__ A, const float* __restrict__ W,
    const float* __restrict__ bias, const float* __restrict__ res,
    void* __restrict__ outp, int M, int N, int K, int flags)
{
  __shared__ _Float16 As[128][40];   // [m][k]
  __shared__ _Float16 Bs[128][40];   // [n][k] (W transposed on stage)

  const int tid = threadIdx.x;
  const int m0 = blockIdx.y * 128, n0 = blockIdx.x * 128;
  const int w  = tid >> 5,  lane = tid & 31;
  const int wm = w >> 1,    wn   = w & 1;
  const int ll = lane & 15, lh   = lane >> 4;

  v8f zf = {0.f,0.f,0.f,0.f,0.f,0.f,0.f,0.f};
  v8f acc[2][4];
  for (int i = 0; i < 2; ++i)
    for (int j = 0; j < 4; ++j) acc[i][j] = zf;

  int aRow[4], aCol[4], wRow[4], wCol[4];
  for (int it = 0; it < 4; ++it) {
    int idx = (it * 256 + tid) * 4;
    aRow[it] = idx >> 5;  aCol[it] = idx & 31;
    wRow[it] = idx >> 7;  wCol[it] = idx & 127;
  }
  v4f aR[4], wR[4];
  for (int it = 0; it < 4; ++it) {
    aR[it] = *(const v4f*)(A + (size_t)(m0 + aRow[it]) * K + aCol[it]);
    wR[it] = *(const v4f*)(W + (size_t)wRow[it] * N + n0 + wCol[it]);
  }

  const int ksteps = K >> 5;
  for (int kt = 0; kt < ksteps; ++kt) {
    __syncthreads();
    for (int it = 0; it < 4; ++it) {
      _Float16* d = &As[aRow[it]][aCol[it]];
      d[0] = (_Float16)aR[it].x; d[1] = (_Float16)aR[it].y;
      d[2] = (_Float16)aR[it].z; d[3] = (_Float16)aR[it].w;
      int nc = wCol[it], kr = wRow[it];
      Bs[nc + 0][kr] = (_Float16)wR[it].x; Bs[nc + 1][kr] = (_Float16)wR[it].y;
      Bs[nc + 2][kr] = (_Float16)wR[it].z; Bs[nc + 3][kr] = (_Float16)wR[it].w;
    }
    __syncthreads();
    if (kt + 1 < ksteps) {
      const int kb2 = (kt + 1) * 32;
      for (int it = 0; it < 4; ++it) {
        aR[it] = *(const v4f*)(A + (size_t)(m0 + aRow[it]) * K + kb2 + aCol[it]);
        wR[it] = *(const v4f*)(W + (size_t)(kb2 + wRow[it]) * N + n0 + wCol[it]);
      }
    }

    v16h af[2], bf[4];
    for (int tm = 0; tm < 2; ++tm) {
      int row = wm * 32 + tm * 16 + ll;
      v8h lo = *(const v8h*)&As[row][lh * 8];
      v8h hi = *(const v8h*)&As[row][lh * 8 + 16];
      af[tm] = cat8(lo, hi);
    }
    for (int tn = 0; tn < 4; ++tn) {
      int col = wn * 64 + tn * 16 + ll;
      v8h lo = *(const v8h*)&Bs[col][lh * 8];
      v8h hi = *(const v8h*)&Bs[col][lh * 8 + 16];
      bf[tn] = cat8(lo, hi);
    }
    for (int tm = 0; tm < 2; ++tm)
      for (int tn = 0; tn < 4; ++tn)
        acc[tm][tn] = __builtin_amdgcn_wmma_f32_16x16x32_f16(
            false, af[tm], false, bf[tn], (short)0, acc[tm][tn], false, false);
  }

  const bool f16o = (flags & GEMM_F16OUT) != 0;
  const bool relu = (flags & GEMM_RELU) != 0;
  for (int tm = 0; tm < 2; ++tm)
    for (int tn = 0; tn < 4; ++tn)
      for (int r = 0; r < 8; ++r) {
        int gr = m0 + wm * 32 + tm * 16 + r + lh * 8;
        int gc = n0 + wn * 64 + tn * 16 + ll;
        float v = acc[tm][tn][r] + bias[gc];
        if (res)  v += res[(size_t)gr * N + gc];
        if (relu) v = v > 0.f ? v : 0.f;
        if (f16o) ((_Float16*)outp)[(size_t)gr * N + gc] = (_Float16)v;
        else      ((float*)outp)[(size_t)gr * N + gc] = v;
      }
}

// ---------------------------------------------------------------------------
// In-place rotary embedding on f16 (B,S,NH*128). 64 threads per (token,head).
// ---------------------------------------------------------------------------
__global__ __launch_bounds__(256) void rope_kernel(_Float16* __restrict__ X)
{
  int hid = blockIdx.x * 4 + (threadIdx.x >> 6);
  int t   = threadIdx.x & 63;
  int pos = (hid / NHc) % Sc;
  _Float16* p = X + (size_t)hid * HDc;
  float x1 = (float)p[2 * t], x2 = (float)p[2 * t + 1];
  float ang = (float)pos * __powf(10000.0f, -(float)t * (1.0f / 64.0f));
  float sn, cs;
  __sincosf(ang, &sn, &cs);
  __syncthreads();
  p[t]      = (_Float16)(x1 * cs - x2 * sn);
  p[64 + t] = (_Float16)(x2 * cs + x1 * sn);
}

// ---------------------------------------------------------------------------
// Flash attention: ctx = softmax(Q Kᵀ * scale + amask) V
// Block = 128 queries x 1 head; 8 waves x 16 rows; key chunks of 64
// (one softmax shuffle-tree / rescale / P-round-trip per 64 keys).
// K/V chunk staging register-pipelined ahead of the WMMAs.
// ---------------------------------------------------------------------------
__global__ __launch_bounds__(256) void flash_wmma(
    const _Float16* __restrict__ Q, const _Float16* __restrict__ Kx,
    const _Float16* __restrict__ V, const float* __restrict__ mask,
    float* __restrict__ ctx)
{
  __shared__ _Float16 Ks[64][136];    // [key][feat]        ~17.4 KB
  __shared__ _Float16 Vs[64][72];     // [col][key]          ~9.2 KB
  __shared__ _Float16 Pp[8][16][72];  // per-wave P patch   ~18.4 KB

  const int tid = threadIdx.x;
  const int w = tid >> 5, lane = tid & 31, ll = lane & 15, lh = lane >> 4;
  const int q0 = blockIdx.x * 128, hh = blockIdx.y, bz = blockIdx.z;
  const float scale = 0.088388347648318447f;  // 1/sqrt(128)

  // Q fragments held in registers across the whole key loop
  v16h aq[4];
  {
    int qrow = q0 + w * 16 + ll;
    const _Float16* qp = Q + ((size_t)(bz * Sc + qrow) * (NHc * HDc) + hh * HDc);
    for (int ks = 0; ks < 4; ++ks) {
      v8h lo = *(const v8h*)(qp + ks * 32 + lh * 8);
      v8h hi = *(const v8h*)(qp + ks * 32 + lh * 8 + 16);
      aq[ks] = cat8(lo, hi);
    }
  }

  v8f zf = {0.f,0.f,0.f,0.f,0.f,0.f,0.f,0.f};
  v8f o[4];
  float mrow[8], lrow[8];
  for (int i = 0; i < 4; ++i) o[i] = zf;
  for (int r = 0; r < 8; ++r) { mrow[r] = -1e30f; lrow[r] = 0.f; }
  const float* mp = mask + (size_t)bz * Sc;

  // staging indices + register pipeline: 4 K-vectors + 2 V-vectors per thread
  const int kKey = tid >> 4, kC8 = tid & 15;   // K: 4 x 16-key groups
  const int vKey = tid >> 3, vCg = tid & 7;    // V: 2 x 32-key groups
  const size_t kBase = (size_t)(bz * Sc) * (NHc * HDc) + hh * HDc + kC8 * 8;
  const size_t vBase = (size_t)(bz * Sc) * (NHc * VDc) + hh * VDc + vCg * 8;
  v8h kR[4], vR[2];
  for (int it = 0; it < 4; ++it)
    kR[it] = *(const v8h*)(Kx + kBase + (size_t)(it * 16 + kKey) * (NHc * HDc));
  for (int it = 0; it < 2; ++it)
    vR[it] = *(const v8h*)(V + vBase + (size_t)(it * 32 + vKey) * (NHc * VDc));

  for (int kb = 0; kb < Sc; kb += 64) {
    __syncthreads();
    // drain staged registers into LDS
    for (int it = 0; it < 4; ++it)
      *(v8h*)&Ks[it * 16 + kKey][kC8 * 8] = kR[it];
    for (int it = 0; it < 2; ++it)
      for (int j = 0; j < 8; ++j)
        Vs[vCg * 8 + j][it * 32 + vKey] = vR[it][j];
    __syncthreads();
    // issue next chunk's global loads before the WMMAs
    if (kb + 64 < Sc) {
      for (int it = 0; it < 4; ++it)
        kR[it] = *(const v8h*)(Kx + kBase +
                               (size_t)(kb + 64 + it * 16 + kKey) * (NHc * HDc));
      for (int it = 0; it < 2; ++it)
        vR[it] = *(const v8h*)(V + vBase +
                               (size_t)(kb + 64 + it * 32 + vKey) * (NHc * VDc));
    }

    // scores: S = Q @ Kᵀ, 4 feature-steps x 4 key tiles
    v8f s[4];
    for (int t = 0; t < 4; ++t) s[t] = zf;
    for (int ks = 0; ks < 4; ++ks) {
      for (int t = 0; t < 4; ++t) {
        v8h lo = *(const v8h*)&Ks[t * 16 + ll][ks * 32 + lh * 8];
        v8h hi = *(const v8h*)&Ks[t * 16 + ll][ks * 32 + lh * 8 + 16];
        v16h bk = cat8(lo, hi);
        s[t] = __builtin_amdgcn_wmma_f32_16x16x32_f16(false, aq[ks], false, bk,
                                                      (short)0, s[t], false, false);
      }
    }

    float am[4];
    for (int t = 0; t < 4; ++t)
      am[t] = (1.0f - mp[kb + t * 16 + ll]) * -10000.0f;

    // online softmax: one shuffle-tree / rescale per 64 keys
    for (int r = 0; r < 8; ++r) {
      float sc[4];
      for (int t = 0; t < 4; ++t) sc[t] = s[t][r] * scale + am[t];
      float cm = fmaxf(fmaxf(sc[0], sc[1]), fmaxf(sc[2], sc[3]));
      cm = fmaxf(cm, __shfl_xor(cm, 1));
      cm = fmaxf(cm, __shfl_xor(cm, 2));
      cm = fmaxf(cm, __shfl_xor(cm, 4));
      cm = fmaxf(cm, __shfl_xor(cm, 8));
      float nm   = fmaxf(mrow[r], cm);
      float corr = __expf(mrow[r] - nm);
      mrow[r] = nm;
      float p[4], rs = 0.f;
      for (int t = 0; t < 4; ++t) { p[t] = __expf(sc[t] - nm); rs += p[t]; }
      rs += __shfl_xor(rs, 1);
      rs += __shfl_xor(rs, 2);
      rs += __shfl_xor(rs, 4);
      rs += __shfl_xor(rs, 8);
      lrow[r] = lrow[r] * corr + rs;
      for (int tn = 0; tn < 4; ++tn) o[tn][r] *= corr;
      int prow = r + lh * 8;
      for (int t = 0; t < 4; ++t)
        Pp[w][prow][t * 16 + ll] = (_Float16)p[t];
    }

    // P (C-layout) -> 2 A-fragments via per-wave LDS patch; PV accumulate
    for (int kc = 0; kc < 2; ++kc) {
      v8h plo = *(const v8h*)&Pp[w][ll][kc * 32 + lh * 8];
      v8h phi = *(const v8h*)&Pp[w][ll][kc * 32 + lh * 8 + 16];
      v16h ap = cat8(plo, phi);
      for (int tn = 0; tn < 4; ++tn) {
        v8h vlo = *(const v8h*)&Vs[tn * 16 + ll][kc * 32 + lh * 8];
        v8h vhi = *(const v8h*)&Vs[tn * 16 + ll][kc * 32 + lh * 8 + 16];
        v16h bv = cat8(vlo, vhi);
        o[tn] = __builtin_amdgcn_wmma_f32_16x16x32_f16(false, ap, false, bv,
                                                       (short)0, o[tn], false, false);
      }
    }
  }

  for (int r = 0; r < 8; ++r) {
    int qg = q0 + w * 16 + r + lh * 8;
    float inv = 1.0f / lrow[r];
    for (int tn = 0; tn < 4; ++tn)
      ctx[(size_t)(bz * Sc + qg) * (NHc * VDc) + hh * VDc + tn * 16 + ll] =
          o[tn][r] * inv;
  }
}

// ---------------------------------------------------------------------------
// LayerNorm over last dim (768). One block per token row.
// ---------------------------------------------------------------------------
__global__ __launch_bounds__(256) void ln_kernel(
    const float* __restrict__ x, const float* __restrict__ g,
    const float* __restrict__ bb, float* __restrict__ y)
{
  __shared__ float red[16];
  const int row = blockIdx.x, tid = threadIdx.x;
  const float* xr = x + (size_t)row * Hc;
  float v0 = xr[tid], v1 = xr[tid + 256], v2 = xr[tid + 512];
  float s = v0 + v1 + v2;
  float q = v0 * v0 + v1 * v1 + v2 * v2;
  for (int off = 1; off < 32; off <<= 1) {
    s += __shfl_xor(s, off);
    q += __shfl_xor(q, off);
  }
  int w = tid >> 5, lane = tid & 31;
  if (lane == 0) { red[w] = s; red[8 + w] = q; }
  __syncthreads();
  float ts = 0.f, tq = 0.f;
  for (int i = 0; i < 8; ++i) { ts += red[i]; tq += red[8 + i]; }
  float u   = ts * (1.0f / Hc);
  float var = tq * (1.0f / Hc) - u * u;
  float inv = rsqrtf(var + 1e-12f);
  float* yr = y + (size_t)row * Hc;
  yr[tid]       = g[tid]       * (v0 - u) * inv + bb[tid];
  yr[tid + 256] = g[tid + 256] * (v1 - u) * inv + bb[tid + 256];
  yr[tid + 512] = g[tid + 512] * (v2 - u) * inv + bb[tid + 512];
}

// ---------------------------------------------------------------------------
extern "C" void kernel_launch(void* const* d_in, const int* in_sizes, int n_in,
                              void* d_out, int out_size, void* d_ws, size_t ws_size,
                              hipStream_t stream)
{
  (void)in_sizes; (void)n_in; (void)out_size; (void)ws_size;
  const float* intent = (const float*)d_in[0];
  const float* slot   = (const float*)d_in[1];
  const float* mask   = (const float*)d_in[2];
  const float* Wq  = (const float*)d_in[3];  const float* bq_  = (const float*)d_in[4];
  const float* Wk  = (const float*)d_in[5];  const float* bk_  = (const float*)d_in[6];
  const float* Wv  = (const float*)d_in[7];  const float* bv_  = (const float*)d_in[8];
  const float* Wqs = (const float*)d_in[9];  const float* bqs  = (const float*)d_in[10];
  const float* Wks = (const float*)d_in[11]; const float* bks  = (const float*)d_in[12];
  const float* Wvs = (const float*)d_in[13]; const float* bvs  = (const float*)d_in[14];
  const float* Wio = (const float*)d_in[15]; const float* bio  = (const float*)d_in[16];
  const float* gio = (const float*)d_in[17]; const float* bbio = (const float*)d_in[18];
  const float* Wso = (const float*)d_in[19]; const float* bso  = (const float*)d_in[20];
  const float* gso = (const float*)d_in[21]; const float* bbso = (const float*)d_in[22];
  const float* Wi1 = (const float*)d_in[23]; const float* bi1  = (const float*)d_in[24];
  const float* Wi2 = (const float*)d_in[25]; const float* bi2  = (const float*)d_in[26];
  const float* gif = (const float*)d_in[27]; const float* bbif = (const float*)d_in[28];
  const float* Ws1 = (const float*)d_in[29]; const float* bs1  = (const float*)d_in[30];
  const float* Ws2 = (const float*)d_in[31]; const float* bs2  = (const float*)d_in[32];
  const float* gsf = (const float*)d_in[33]; const float* bsf  = (const float*)d_in[34];

  char* ws = (char*)d_ws;
  size_t off = 0;
  auto alloc = [&](size_t bytes) -> char* {
    char* p = ws + off;
    off += (bytes + 255) & ~(size_t)255;
    return p;
  };
  _Float16* Qi   = (_Float16*)alloc((size_t)Mc * 1536 * 2);
  _Float16* Ki   = (_Float16*)alloc((size_t)Mc * 1536 * 2);
  _Float16* Qs   = (_Float16*)alloc((size_t)Mc * 1536 * 2);
  _Float16* Ks_  = (_Float16*)alloc((size_t)Mc * 1536 * 2);
  _Float16* Vi   = (_Float16*)alloc((size_t)Mc * 768 * 2);
  _Float16* Vs_  = (_Float16*)alloc((size_t)Mc * 768 * 2);
  float* ctx_i   = (float*)alloc((size_t)Mc * 768 * 4);
  float* ctx_s   = (float*)alloc((size_t)Mc * 768 * 4);
  float* tmp     = (float*)alloc((size_t)Mc * 768 * 4);
  float* hbuf    = (float*)alloc((size_t)Mc * 768 * 4);
  float* act     = (float*)alloc((size_t)Mc * 3072 * 4);

  float* out_i = (float*)d_out;
  float* out_s = (float*)d_out + (size_t)Mc * Hc;

  const dim3 blk(256);
  const dim3 gQK(1536 / 128, Mc / 128);
  const dim3 gV(768 / 128, Mc / 128);
  const dim3 gF1(3072 / 128, Mc / 128);

  // 1) QKV projections (f16 out)
  gemm_wmma<<<gQK, blk, 0, stream>>>(intent, Wq,  bq_, nullptr, Qi,  Mc, 1536, 768, GEMM_F16OUT);
  gemm_wmma<<<gQK, blk, 0, stream>>>(slot,   Wk,  bk_, nullptr, Ki,  Mc, 1536, 768, GEMM_F16OUT);
  gemm_wmma<<<gV,  blk, 0, stream>>>(slot,   Wv,  bv_, nullptr, Vi,  Mc, 768,  768, GEMM_F16OUT);
  gemm_wmma<<<gQK, blk, 0, stream>>>(slot,   Wqs, bqs, nullptr, Qs,  Mc, 1536, 768, GEMM_F16OUT);
  gemm_wmma<<<gQK, blk, 0, stream>>>(intent, Wks, bks, nullptr, Ks_, Mc, 1536, 768, GEMM_F16OUT);
  gemm_wmma<<<gV,  blk, 0, stream>>>(intent, Wvs, bvs, nullptr, Vs_, Mc, 768,  768, GEMM_F16OUT);

  // 2) RoPE (in place)
  const int rope_blocks = Bc * Sc * NHc / 4;
  rope_kernel<<<rope_blocks, blk, 0, stream>>>(Qi);
  rope_kernel<<<rope_blocks, blk, 0, stream>>>(Ki);
  rope_kernel<<<rope_blocks, blk, 0, stream>>>(Qs);
  rope_kernel<<<rope_blocks, blk, 0, stream>>>(Ks_);

  // 3) flash attention (both streams)
  const dim3 gFA(Sc / 128, NHc, Bc);
  flash_wmma<<<gFA, blk, 0, stream>>>(Qi, Ki, Vi, mask, ctx_i);
  flash_wmma<<<gFA, blk, 0, stream>>>(Qs, Ks_, Vs_, mask, ctx_s);

  // 4) intent chain: selfout -> LN -> FFN -> LN -> d_out[0..]
  gemm_wmma<<<gV, blk, 0, stream>>>(ctx_i, Wio, bio, intent, tmp, Mc, 768, 768, 0);
  ln_kernel<<<Mc, blk, 0, stream>>>(tmp, gio, bbio, hbuf);
  gemm_wmma<<<gF1, blk, 0, stream>>>(hbuf, Wi1, bi1, nullptr, act, Mc, 3072, 768, GEMM_RELU);
  gemm_wmma<<<gV, blk, 0, stream>>>(act, Wi2, bi2, hbuf, tmp, Mc, 768, 3072, 0);
  ln_kernel<<<Mc, blk, 0, stream>>>(tmp, gif, bbif, out_i);

  // 5) slot chain: selfout -> LN -> FFN -> LN -> d_out[B*S*H..]
  gemm_wmma<<<gV, blk, 0, stream>>>(ctx_s, Wso, bso, slot, tmp, Mc, 768, 768, 0);
  ln_kernel<<<Mc, blk, 0, stream>>>(tmp, gso, bbso, hbuf);
  gemm_wmma<<<gF1, blk, 0, stream>>>(hbuf, Ws1, bs1, nullptr, act, Mc, 3072, 768, GEMM_RELU);
  gemm_wmma<<<gV, blk, 0, stream>>>(act, Ws2, bs2, hbuf, tmp, Mc, 768, 3072, 0);
  ln_kernel<<<Mc, blk, 0, stream>>>(tmp, gsf, bsf, out_s);
}
